// SparseAutoencoder_1812476199461
// MI455X (gfx1250) — compile-verified, matrix operand order checked by reading, not run on hardware
//
#include <hip/hip_runtime.h>
#include <stdint.h>

#define N_TOKENS   8192
#define INPUT_DIM  2048
#define HIDDEN_DIM 16384
#define K_TOP      64

typedef __attribute__((ext_vector_type(16))) __bf16 v16bf;
typedef __attribute__((ext_vector_type(8)))  float  v8f;

// ---------------- helpers ----------------
__device__ __forceinline__ unsigned short f2bf(float f) {
  unsigned int u = __float_as_uint(f);
  unsigned int r = u + 0x7FFFu + ((u >> 16) & 1u);  // round-to-nearest-even
  return (unsigned short)(r >> 16);
}

// async global -> LDS copy (CDNA5), 16B per lane; tracked by ASYNCcnt
#define ASYNC_COPY_B128(ldsOff, gAddr)                                        \
  asm volatile("global_load_async_to_lds_b128 %0, %1, off"                    \
               :: "v"(ldsOff), "v"(gAddr) : "memory")
#define WAIT_ASYNC0() asm volatile("s_wait_asynccnt 0" ::: "memory")

// ---------------- kernel 1a: center x and cast to bf16 (packed pairs) ----------------
__global__ void sae_center_cast_x(const float* __restrict__ x,
                                  const float* __restrict__ mean,
                                  uint32_t* __restrict__ xbf) {
  const int64_t total = (int64_t)N_TOKENS * INPUT_DIM / 2;  // pairs
  for (int64_t p = blockIdx.x * (int64_t)blockDim.x + threadIdx.x; p < total;
       p += (int64_t)gridDim.x * blockDim.x) {
    float2 f = ((const float2*)x)[p];
    int d0 = (int)((p * 2) & (INPUT_DIM - 1));
    float a = f.x - mean[d0];
    float b = f.y - mean[d0 + 1];
    xbf[p] = (uint32_t)f2bf(a) | ((uint32_t)f2bf(b) << 16);
  }
}

// ---------------- kernel 1b: cast enc_w to bf16 ----------------
__global__ void sae_cast_w(const float* __restrict__ w, uint32_t* __restrict__ wbf) {
  const int64_t total = (int64_t)HIDDEN_DIM * INPUT_DIM / 2;
  for (int64_t p = blockIdx.x * (int64_t)blockDim.x + threadIdx.x; p < total;
       p += (int64_t)gridDim.x * blockDim.x) {
    float2 f = ((const float2*)w)[p];
    wbf[p] = (uint32_t)f2bf(f.x) | ((uint32_t)f2bf(f.y) << 16);
  }
}

// ---------------- kernel 2: transpose dec_w [2048,16384] -> [16384,2048] ----------------
__global__ __launch_bounds__(256)
void sae_transpose_w(const float* __restrict__ src, float* __restrict__ dst) {
  __shared__ float tile[32][33];
  const int tx = threadIdx.x & 31;
  const int ty = threadIdx.x >> 5;              // 0..7
  const int xb = blockIdx.x * 32;               // along HIDDEN_DIM
  const int yb = blockIdx.y * 32;               // along INPUT_DIM
#pragma unroll
  for (int j = 0; j < 4; ++j) {
    int y = yb + ty + j * 8;
    tile[ty + j * 8][tx] = src[(int64_t)y * HIDDEN_DIM + xb + tx];
  }
  __syncthreads();
#pragma unroll
  for (int j = 0; j < 4; ++j) {
    int r = ty + j * 8;
    dst[(int64_t)(xb + r) * INPUT_DIM + yb + tx] = tile[tx][r];
  }
}

// ---------------- kernel 3: encoder GEMM z = xbf @ wbf^T + enc_b (bf16 WMMA) ----------
// Block tile 128x128, BK=32, 8 waves, wave tile 32x64 -> 8 WMMAs / wave / K-step.
// Double-buffered LDS fed by global_load_async_to_lds_b128; one barrier per K-step.
#define BM 128
#define BN 128
#define BK 32
#define LP 40                       // LDS row pitch (ushort): 80B, non-pow2 bank spread
#define TILE_ELEMS (BM * LP)

__global__ __launch_bounds__(256)
void sae_encode_gemm(const uint32_t* __restrict__ xbf,  // [8192][2048] bf16 pairs
                     const uint32_t* __restrict__ wbf,  // [16384][2048] bf16 pairs
                     const float* __restrict__ enc_b,
                     float* __restrict__ z) {
  __shared__ alignas(16) unsigned short As[2][TILE_ELEMS];
  __shared__ alignas(16) unsigned short Ws[2][TILE_ELEMS];

  const int tid  = threadIdx.x;
  const int lane = tid & 31;
  const int wave = tid >> 5;                 // 0..7
  const int m0 = blockIdx.y * BM;
  const int n0 = blockIdx.x * BN;
  const int wm = (wave & 3) * 32;            // wave M offset in block tile
  const int wn = (wave >> 2) * 64;           // wave N offset in block tile

  // cooperative async load mapping: 128 rows x 4 chunks(16B) per matrix;
  // thread handles rows (tid>>2) and (tid>>2)+64, chunk tid&3.
  const int lrow = tid >> 2;                 // 0..63
  const int lchk = tid & 3;                  // 0..3
  const size_t ROWB = INPUT_DIM * 2;         // 4096B per bf16 row

  uint64_t aAddr0 = (uint64_t)(uintptr_t)((const char*)xbf + (size_t)(m0 + lrow) * ROWB + lchk * 16);
  uint64_t aAddr1 = aAddr0 + 64ull * ROWB;
  uint64_t wAddr0 = (uint64_t)(uintptr_t)((const char*)wbf + (size_t)(n0 + lrow) * ROWB + lchk * 16);
  uint64_t wAddr1 = wAddr0 + 64ull * ROWB;

  uint32_t ldsA0[2], ldsA1[2], ldsW0[2], ldsW1[2];
#pragma unroll
  for (int b = 0; b < 2; ++b) {
    ldsA0[b] = (uint32_t)(uintptr_t)&As[b][lrow * LP + lchk * 8];
    ldsA1[b] = (uint32_t)(uintptr_t)&As[b][(lrow + 64) * LP + lchk * 8];
    ldsW0[b] = (uint32_t)(uintptr_t)&Ws[b][lrow * LP + lchk * 8];
    ldsW1[b] = (uint32_t)(uintptr_t)&Ws[b][(lrow + 64) * LP + lchk * 8];
  }

  v8f acc[2][4];
#pragma unroll
  for (int mh = 0; mh < 2; ++mh)
#pragma unroll
    for (int nf = 0; nf < 4; ++nf)
      acc[mh][nf] = (v8f){0.f, 0.f, 0.f, 0.f, 0.f, 0.f, 0.f, 0.f};

  // prologue: async-fill buffer 0 with K tile 0
  ASYNC_COPY_B128(ldsA0[0], aAddr0);
  ASYNC_COPY_B128(ldsA1[0], aAddr1);
  ASYNC_COPY_B128(ldsW0[0], wAddr0);
  ASYNC_COPY_B128(ldsW1[0], wAddr1);
  aAddr0 += 64; aAddr1 += 64; wAddr0 += 64; wAddr1 += 64;   // BK=32 bf16 = 64B
  WAIT_ASYNC0();
  __syncthreads();

  // per-lane fragment addressing (16-bit A 16x32 layout):
  // lanes 0-15: v0..3=K0..7, v4..7=K16..23 ; lanes 16-31: K8..15 / K24..31
  const int fl   = lane & 15;
  const int ksel = (lane >> 4) * 8;          // element offset 0 or 8

  int cur = 0;
  for (int k0 = 0; k0 < INPUT_DIM; k0 += BK) {
    const int nxt = cur ^ 1;
    const bool hasNext = (k0 + BK) < INPUT_DIM;
    if (hasNext) {                            // async-fill next buffer
      ASYNC_COPY_B128(ldsA0[nxt], aAddr0);
      ASYNC_COPY_B128(ldsA1[nxt], aAddr1);
      ASYNC_COPY_B128(ldsW0[nxt], wAddr0);
      ASYNC_COPY_B128(ldsW1[nxt], wAddr1);
      aAddr0 += 64; aAddr1 += 64; wAddr0 += 64; wAddr1 += 64;
    }

    // load fragments from current buffer and issue 8 WMMAs
    union Frag { uint4 u[2]; v16bf v; };
    Frag fa[2], fb[4];
#pragma unroll
    for (int mh = 0; mh < 2; ++mh) {
      const unsigned short* p = &As[cur][(wm + mh * 16 + fl) * LP + ksel];
      fa[mh].u[0] = *(const uint4*)(p);
      fa[mh].u[1] = *(const uint4*)(p + 16);
    }
#pragma unroll
    for (int nf = 0; nf < 4; ++nf) {
      const unsigned short* p = &Ws[cur][(wn + nf * 16 + fl) * LP + ksel];
      fb[nf].u[0] = *(const uint4*)(p);
      fb[nf].u[1] = *(const uint4*)(p + 16);
    }
#pragma unroll
    for (int mh = 0; mh < 2; ++mh)
#pragma unroll
      for (int nf = 0; nf < 4; ++nf)
        acc[mh][nf] = __builtin_amdgcn_wmma_f32_16x16x32_bf16(
            false, fa[mh].v, false, fb[nf].v, (short)0, acc[mh][nf], false, false);

    WAIT_ASYNC0();       // my async fills of nxt are in LDS
    __syncthreads();     // everyone done reading cur & filling nxt
    cur = nxt;
  }

  // store: C layout VGPR r -> M = r (+8 for lanes>=16), N = lane&15
  const int mhalf = (lane >> 4) ? 8 : 0;
#pragma unroll
  for (int mh = 0; mh < 2; ++mh) {
#pragma unroll
    for (int nf = 0; nf < 4; ++nf) {
      const int n = n0 + wn + nf * 16 + fl;
      const float bias = enc_b[n];
      const int mb = m0 + wm + mh * 16 + mhalf;
#pragma unroll
      for (int r = 0; r < 8; ++r) {
        __builtin_nontemporal_store(acc[mh][nf][r] + bias,
                                    &z[(int64_t)(mb + r) * HIDDEN_DIM + n]);
      }
    }
  }
}

// ---------------- kernel 4: per-row top-64 ----------------
__global__ __launch_bounds__(256)
void sae_topk(const float* __restrict__ z, float* __restrict__ topv, int* __restrict__ topi) {
  __shared__ float srow[HIDDEN_DIM];   // 64 KB row cache (320 KB/WGP budget)
  __shared__ float rv[8];
  __shared__ int   ri[8];
  __shared__ float ov[K_TOP];
  __shared__ int   oi[K_TOP];

  const int row = blockIdx.x;
  const int tid = threadIdx.x;
  const int lane = tid & 31;
  const int wave = tid >> 5;

  const float* zr = z + (int64_t)row * HIDDEN_DIM;
  for (int i = tid; i < HIDDEN_DIM; i += 256)
    srow[i] = __builtin_nontemporal_load(&zr[i]);   // z is stream-once
  __syncthreads();

  for (int it = 0; it < K_TOP; ++it) {
    float best = -__builtin_huge_valf();
    int   bi   = 0;
    for (int i = tid; i < HIDDEN_DIM; i += 256) {
      float v = srow[i];
      if (v > best) { best = v; bi = i; }
    }
#pragma unroll
    for (int off = 16; off > 0; off >>= 1) {        // wave32 butterfly
      float vv = __shfl_xor(best, off, 32);
      int   ii = __shfl_xor(bi,   off, 32);
      if (vv > best) { best = vv; bi = ii; }
    }
    if (lane == 0) { rv[wave] = best; ri[wave] = bi; }
    __syncthreads();
    if (tid == 0) {
      float b = rv[0]; int j = ri[0];
#pragma unroll
      for (int w = 1; w < 8; ++w)
        if (rv[w] > b) { b = rv[w]; j = ri[w]; }
      ov[it] = b; oi[it] = j;
      srow[j] = -__builtin_huge_valf();
    }
    __syncthreads();
  }
  if (tid < K_TOP) {
    topv[row * K_TOP + tid] = ov[tid];
    topi[row * K_TOP + tid] = oi[tid];
  }
}

// ---------------- kernel 5: sparse decode ----------------
__global__ __launch_bounds__(256)
void sae_decode(const float* __restrict__ topv, const int* __restrict__ topi,
                const float* __restrict__ dec_wT,  // [16384][2048]
                const float* __restrict__ dec_b, const float* __restrict__ mean,
                float* __restrict__ out) {
  __shared__ float sv[K_TOP];
  __shared__ int   si[K_TOP];
  const int row = blockIdx.x;
  const int tid = threadIdx.x;
  if (tid < K_TOP) {
    sv[tid] = topv[row * K_TOP + tid];
    si[tid] = topi[row * K_TOP + tid];
  }
  __syncthreads();

  float acc[INPUT_DIM / 256];   // 8 outputs per thread
#pragma unroll
  for (int j = 0; j < 8; ++j) {
    int d = tid + j * 256;
    acc[j] = dec_b[d] + mean[d];
  }
  for (int k = 0; k < K_TOP; ++k) {
    const float v = sv[k];
    const float* wr = dec_wT + (int64_t)si[k] * INPUT_DIM;
#pragma unroll
    for (int j = 0; j < 8; ++j) acc[j] += v * wr[tid + j * 256];  // coalesced
  }
#pragma unroll
  for (int j = 0; j < 8; ++j)
    out[(int64_t)row * INPUT_DIM + tid + j * 256] = acc[j];
}

// ---------------- launch ----------------
extern "C" void kernel_launch(void* const* d_in, const int* in_sizes, int n_in,
                              void* d_out, int out_size, void* d_ws, size_t ws_size,
                              hipStream_t stream) {
  const float* x     = (const float*)d_in[0];
  const float* enc_w = (const float*)d_in[1];
  const float* enc_b = (const float*)d_in[2];
  const float* dec_w = (const float*)d_in[3];
  const float* dec_b = (const float*)d_in[4];
  const float* mean  = (const float*)d_in[5];
  float* out = (float*)d_out;

  // workspace layout
  char* ws = (char*)d_ws;
  const size_t Z_BYTES   = (size_t)N_TOKENS * HIDDEN_DIM * 4;   // 512 MB
  const size_t XBF_BYTES = (size_t)N_TOKENS * INPUT_DIM * 2;    //  32 MB
  const size_t WBF_BYTES = (size_t)HIDDEN_DIM * INPUT_DIM * 2;  //  64 MB
  const size_t WT_BYTES  = (size_t)HIDDEN_DIM * INPUT_DIM * 4;  // 128 MB
  const size_t TV_BYTES  = (size_t)N_TOKENS * K_TOP * 4;        //   2 MB

  float*    z    = (float*)ws;
  uint32_t* xbf  = (uint32_t*)(ws + Z_BYTES);
  uint32_t* wbf  = (uint32_t*)(ws + Z_BYTES + XBF_BYTES);
  float*    wT   = (float*)(ws + Z_BYTES + XBF_BYTES + WBF_BYTES);
  float*    topv = (float*)(ws + Z_BYTES + XBF_BYTES + WBF_BYTES + WT_BYTES);
  int*      topi = (int*)(ws + Z_BYTES + XBF_BYTES + WBF_BYTES + WT_BYTES + TV_BYTES);

  sae_center_cast_x<<<4096, 256, 0, stream>>>(x, mean, xbf);
  sae_cast_w<<<8192, 256, 0, stream>>>(enc_w, wbf);
  sae_transpose_w<<<dim3(HIDDEN_DIM / 32, INPUT_DIM / 32), 256, 0, stream>>>(dec_w, wT);
  sae_encode_gemm<<<dim3(HIDDEN_DIM / BN, N_TOKENS / BM), 256, 0, stream>>>(xbf, wbf, enc_b, z);
  sae_topk<<<N_TOKENS, 256, 0, stream>>>(z, topv, topi);
  sae_decode<<<N_TOKENS, 256, 0, stream>>>(topv, topi, wT, dec_b, mean, out);
}
